// FactoredGNNPolicy_12352325943982
// MI455X (gfx1250) — matrix operand order repaired and checked by reading, not compiled, not run on hardware
//
#include <hip/hip_runtime.h>
#include <hip/hip_bf16.h>

typedef __attribute__((ext_vector_type(16))) __bf16 v16bf;
typedef __attribute__((ext_vector_type(8)))  float  v8f;

// ---------------------------------------------------------------------------
// float atomic max via monotone int/uint punning. Sentinel = 0xFFFFFFFF bytes
// (memset 0xFF): as int it is -1 (any non-negative float beats it via int max),
// as uint it is UINT_MAX (any negative float beats it via uint min).
// ---------------------------------------------------------------------------
__device__ __forceinline__ void atomicMaxFloat(float* addr, float val) {
  if (val >= 0.0f) atomicMax((int*)addr, __float_as_int(val));
  else             atomicMin((unsigned int*)addr, __float_as_uint(val));
}

// ---------------------------------------------------------------------------
// Shared-memory W staging in *WMMA B-fragment order*:
// fragment (t = col tile 0..3, u = k-step 0..1), lane 0..31, elem 0..15
//   element value = W[k][n],  k = u*32 + (e>>3)*16 + (lane>>4)*8 + (e&7),  n = t*16 + (lane&15)
// so each lane's 16 bf16 fragment is one aligned 32B LDS read (2x ds_load_b128).
// ---------------------------------------------------------------------------
template<int K>
__device__ __forceinline__ void stage_W(const float* __restrict__ W, __bf16* sWf, int tid) {
#pragma unroll
  for (int i = tid; i < 4096; i += 256) {
    int e  = i & 15;
    int ln = (i >> 4) & 31;
    int u  = (i >> 9) & 1;
    int t  = (i >> 10);
    int k  = u * 32 + (e >> 3) * 16 + (ln >> 4) * 8 + (e & 7);
    int n  = t * 16 + (ln & 15);
    sWf[i] = (__bf16)((k < K) ? W[k * 64 + n] : 0.0f);
  }
}

// Load 16x32 bf16 A fragments for one wave (rows row0..row0+15), K multiple of 16.
template<int K>
__device__ __forceinline__ void load_A_frags(const float* __restrict__ arow, int hh, v16bf af[2]) {
#pragma unroll
  for (int u = 0; u < 2; ++u) {
#pragma unroll
    for (int h2 = 0; h2 < 2; ++h2) {           // lower/upper 8 elems of fragment
      const int kchunk = u * 32 + h2 * 16;     // compile-time chunk base
      if (kchunk < K) {
        const float* p = arow + kchunk + hh * 8;
        float4 x0 = *(const float4*)(p);
        float4 x1 = *(const float4*)(p + 4);
        af[u][h2 * 8 + 0] = (__bf16)x0.x; af[u][h2 * 8 + 1] = (__bf16)x0.y;
        af[u][h2 * 8 + 2] = (__bf16)x0.z; af[u][h2 * 8 + 3] = (__bf16)x0.w;
        af[u][h2 * 8 + 4] = (__bf16)x1.x; af[u][h2 * 8 + 5] = (__bf16)x1.y;
        af[u][h2 * 8 + 6] = (__bf16)x1.z; af[u][h2 * 8 + 7] = (__bf16)x1.w;
      } else {
#pragma unroll
        for (int e2 = 0; e2 < 8; ++e2) af[u][h2 * 8 + e2] = (__bf16)0.0f;
      }
    }
  }
}

// ---------------------------------------------------------------------------
// GEMM: out[N,64] = act(A[N,K] @ W[K,64] + bias), bf16 WMMA, f32 accumulate.
// Block = 256 thr = 8 waves; wave -> 16 rows x 64 cols (4 col tiles x 2 k-steps)
// ---------------------------------------------------------------------------
template<int K, int RELU>
__global__ void gemm_n64(const float* __restrict__ A, const float* __restrict__ W,
                         const float* __restrict__ bias, float* __restrict__ out, int N)
{
  __shared__ __attribute__((aligned(32))) __bf16 sWf[4096];
  __shared__ float sB[64];
  int tid = threadIdx.x;
  stage_W<K>(W, sWf, tid);
  if (tid < 64) sB[tid] = bias[tid];
  __syncthreads();

  int wave = tid >> 5, lane = tid & 31;
  int m = lane & 15, hh = lane >> 4;
  long row0 = ((long)blockIdx.x * 8 + wave) * 16;
  if (row0 >= N) return;

  v16bf af[2];
  load_A_frags<K>(A + (row0 + m) * (long)K, hh, af);

  const v16bf* fr = (const v16bf*)sWf;
#pragma unroll
  for (int t = 0; t < 4; ++t) {
    v16bf b0 = fr[(t * 2 + 0) * 32 + lane];
    v16bf b1 = fr[(t * 2 + 1) * 32 + lane];
    v8f acc = {};
    acc = __builtin_amdgcn_wmma_f32_16x16x32_bf16(false, af[0], false, b0, (short)0, acc, false, false);
    acc = __builtin_amdgcn_wmma_f32_16x16x32_bf16(false, af[1], false, b1, (short)0, acc, false, false);
#pragma unroll
    for (int r = 0; r < 8; ++r) {
      float v = acc[r] + sB[t * 16 + m];
      if (RELU) v = fmaxf(v, 0.0f);
      out[(row0 + hh * 8 + r) * 64 + t * 16 + m] = v;
    }
  }
}

// ---------------------------------------------------------------------------
// Fused head: score[i] = relu(A[i,:] @ W1[0:64,:] + c) . w2 + b2
// Same WMMA tiling; @w2 fused in epilogue with 16-lane shuffle reduce.
// ---------------------------------------------------------------------------
__global__ void head_score(const float* __restrict__ A, const float* __restrict__ W1,
                           const float* __restrict__ cvec, const float* __restrict__ w2,
                           const float* __restrict__ b2, float* __restrict__ out, int N)
{
  __shared__ __attribute__((aligned(32))) __bf16 sWf[4096];
  __shared__ float sc[64], sw2[64];
  int tid = threadIdx.x;
  stage_W<64>(W1, sWf, tid);
  if (tid < 64) { sc[tid] = cvec[tid]; sw2[tid] = w2[tid]; }
  __syncthreads();

  int wave = tid >> 5, lane = tid & 31;
  int m = lane & 15, hh = lane >> 4;
  long row0 = ((long)blockIdx.x * 8 + wave) * 16;
  if (row0 >= N) return;

  v16bf af[2];
  load_A_frags<64>(A + (row0 + m) * 64L, hh, af);

  const v16bf* fr = (const v16bf*)sWf;
  float p[8] = {0, 0, 0, 0, 0, 0, 0, 0};
#pragma unroll
  for (int t = 0; t < 4; ++t) {
    v16bf b0 = fr[(t * 2 + 0) * 32 + lane];
    v16bf b1 = fr[(t * 2 + 1) * 32 + lane];
    v8f acc = {};
    acc = __builtin_amdgcn_wmma_f32_16x16x32_bf16(false, af[0], false, b0, (short)0, acc, false, false);
    acc = __builtin_amdgcn_wmma_f32_16x16x32_bf16(false, af[1], false, b1, (short)0, acc, false, false);
#pragma unroll
    for (int r = 0; r < 8; ++r) {
      float x = fmaxf(acc[r] + sc[t * 16 + m], 0.0f);
      p[r] += x * sw2[t * 16 + m];
    }
  }
#pragma unroll
  for (int off = 1; off < 16; off <<= 1)
#pragma unroll
    for (int r = 0; r < 8; ++r) p[r] += __shfl_xor(p[r], off, 32);
  if (m == 0) {
    float bb = b2[0];
#pragma unroll
    for (int r = 0; r < 8; ++r) out[row0 + hh * 8 + r] = p[r] + bb;
  }
}

// ---------------------------------------------------------------------------
// Edge kernels (memory-bound, sub-wave per edge)
// ---------------------------------------------------------------------------
__global__ void edge_logits(const float* __restrict__ xl, const float* __restrict__ xr,
                            const int* __restrict__ ei, const float* __restrict__ att,
                            float* __restrict__ ebuf, float* __restrict__ emax, int E_)
{
  long gid = (long)blockIdx.x * 256 + threadIdx.x;
  long eid = gid >> 3;
  if (eid >= E_) return;
  int j = (int)(gid & 7);
  int s = ei[eid], d = ei[(long)E_ + eid];
  const float4* pl = (const float4*)(xl + (long)s * 64 + j * 8);
  const float4* pr = (const float4*)(xr + (long)d * 64 + j * 8);
  float part = 0.0f;
#pragma unroll
  for (int q = 0; q < 2; ++q) {
    float4 a = pl[q], b = pr[q];
    const float* ap = (const float*)&a;
    const float* bp = (const float*)&b;
#pragma unroll
    for (int c2 = 0; c2 < 4; ++c2) {
      float x = ap[c2] + bp[c2];
      x = x > 0.0f ? x : 0.2f * x;          // leaky_relu(0.2)
      part += x * att[j * 8 + q * 4 + c2];
    }
  }
  for (int off = 1; off < 8; off <<= 1) part += __shfl_xor(part, off, 32);
  if (j == 0) { ebuf[eid] = part; atomicMaxFloat(&emax[d], part); }
}

__global__ void edge_softmax_num(const float* __restrict__ ebuf, const float* __restrict__ emax,
                                 const int* __restrict__ ei, float* __restrict__ wbuf,
                                 float* __restrict__ wsum, int E_)
{
  long i = (long)blockIdx.x * 256 + threadIdx.x;
  if (i >= E_) return;
  int d = ei[(long)E_ + i];
  float wv = __expf(ebuf[i] - emax[d]);
  wbuf[i] = wv;
  atomicAdd(&wsum[d], wv);
}

__global__ void edge_scatter(const float* __restrict__ xl, const int* __restrict__ ei,
                             const float* __restrict__ wbuf, const float* __restrict__ wsum,
                             float* __restrict__ outbuf, int E_)
{
  long gid = (long)blockIdx.x * 256 + threadIdx.x;
  long eid = gid >> 4;
  if (eid >= E_) return;
  int j = (int)(gid & 15);
  int s = ei[eid], d = ei[(long)E_ + eid];
  float alpha = wbuf[eid] / wsum[d];
  float4 x = *(const float4*)(xl + (long)s * 64 + j * 4);
  float* o = outbuf + (long)d * 64 + j * 4;
  atomicAdd(o + 0, alpha * x.x);
  atomicAdd(o + 1, alpha * x.y);
  atomicAdd(o + 2, alpha * x.z);
  atomicAdd(o + 3, alpha * x.w);
}

// ---------------------------------------------------------------------------
// Small helper kernels
// ---------------------------------------------------------------------------
__global__ void vec_accum3(const float* a, const float* b, const float* c, float* dst) {
  int j = threadIdx.x;
  float v = a[j];
  if (b) v += b[j];
  if (c) v += c[j];
  dst[j] = v;
}

__global__ void add_bias_relu(const float* __restrict__ in, const float* __restrict__ bvec,
                              float* __restrict__ out, long n)
{
  long i = (long)blockIdx.x * 256 + threadIdx.x;
  if (i >= n) return;
  out[i] = fmaxf(in[i] + bvec[i & 63], 0.0f);
}

__global__ void colsum(const float* __restrict__ h, int N, float* __restrict__ sums) {
  int col = threadIdx.x & 63;
  int rg  = threadIdx.x >> 6;   // 4 rows per block pass
  float local = 0.0f;
  for (long r = (long)blockIdx.x * 4 + rg; r < N; r += (long)gridDim.x * 4)
    local += h[r * 64 + col];
  atomicAdd(&sums[col], local);
}

__global__ void finalize_g(const float* sums, const float* type_emb, const int* tidx,
                           float* g, float* temb, int Nn, int Nh, int Ns, int Nd)
{
  int c = threadIdx.x;
  g[c] = 0.25f * (sums[c] / (float)Nn + sums[64 + c] / (float)Nh +
                  sums[128 + c] / (float)Ns + sums[192 + c] / (float)Nd);
  temb[c] = type_emb[tidx[0] * 64 + c];
}

__global__ void ctx_heads(const float* __restrict__ g, const float* __restrict__ ss,
                          const float* tW1, const float* tW2, const float* tb1, const float* tb2,
                          const float* vW1, const float* vW2, const float* vb1, const float* vb2,
                          float* __restrict__ out)
{
  __shared__ float ctx[67], h1[64], h2[64];
  int j = threadIdx.x;
  if (j < 64) ctx[j] = g[j];
  if (j < 3)  ctx[64 + j] = ss[j];
  __syncthreads();
  if (j < 64) {
    float a1 = tb1[j], a2 = vb1[j];
    for (int k = 0; k < 67; ++k) { a1 += ctx[k] * tW1[k * 64 + j]; a2 += ctx[k] * vW1[k * 64 + j]; }
    h1[j] = fmaxf(a1, 0.0f);
    h2[j] = fmaxf(a2, 0.0f);
  }
  __syncthreads();
  if (j < 5) {
    float a = tb2[j];
    for (int k = 0; k < 64; ++k) a += h1[k] * tW2[k * 5 + j];
    out[j] = a;
  }
  if (j == 5) {
    float a = vb2[0];
    for (int k = 0; k < 64; ++k) a += h2[k] * vW2[k];
    out[5] = a;
  }
}

// c[j] = b1[j] + g.W1[64:128] + temb.W1[128:192] (+ semb.W1[192:256]) (+ demb.W1[256:320])
__global__ void cbias(const float* W1, const float* b1, const float* g, const float* temb,
                      const float* semb, const float* demb, float* c)
{
  int j = threadIdx.x;
  float a = b1[j];
  for (int k = 0; k < 64; ++k) a += g[k]    * W1[(64  + k) * 64 + j];
  for (int k = 0; k < 64; ++k) a += temb[k] * W1[(128 + k) * 64 + j];
  if (semb) for (int k = 0; k < 64; ++k) a += semb[k] * W1[(192 + k) * 64 + j];
  if (demb) for (int k = 0; k < 64; ++k) a += demb[k] * W1[(256 + k) * 64 + j];
  c[j] = a;
}

// ---------------------------------------------------------------------------
// Orchestration
// ---------------------------------------------------------------------------
extern "C" void kernel_launch(void* const* d_in, const int* in_sizes, int n_in,
                              void* d_out, int out_size, void* d_ws, size_t ws_size,
                              hipStream_t stream)
{
  (void)n_in; (void)out_size; (void)ws_size;

  // node types: 0=network 1=host 2=service 3=data
  const float* X[4] = {(const float*)d_in[0], (const float*)d_in[1],
                       (const float*)d_in[2], (const float*)d_in[3]};
  int NSZ[4] = {in_sizes[0] / 16, in_sizes[1] / 16, in_sizes[2] / 16, in_sizes[3] / 16};
  int Ecnt = in_sizes[4] / 2;

  // params pytree (sorted dict keys): convs[0](36), convs[1](36), enc(8),
  // sec_head(4), src_head(4), tgt_head(4), type_emb(1), type_head(4), value_head(4)
  static const int EG_EI[6]  = {9, 4, 6, 7, 5, 8}; // input index of edge_index
  static const int EG_SRC[6] = {3, 1, 1, 1, 0, 2};
  static const int EG_DST[6] = {1, 0, 2, 3, 1, 1};
  static const int ENC_ORD[4] = {2, 1, 3, 0};      // node type -> position in sorted enc dict
  const int PCONV = 10, PENC = 82, PSEC = 90, PSRC = 94, PTGT = 98,
            PTEMB = 102, PTYPE = 103, PVAL = 107, PSS = 111, PTIDX = 112;

  float* ws = (float*)d_ws;
  long HOFF[4]; HOFF[0] = 0;
  for (int t = 1; t < 4; ++t) HOFF[t] = HOFF[t - 1] + (long)NSZ[t - 1] * 64;
  long HT = HOFF[3] + (long)NSZ[3] * 64;
  int NMAX = NSZ[0];
  for (int t = 1; t < 4; ++t) if (NSZ[t] > NMAX) NMAX = NSZ[t];

  float* hbuf  = ws;
  float* nbuf  = hbuf + HT;
  float* xlbuf = nbuf + HT;
  float* xrbuf = xlbuf + (long)NMAX * 64;
  float* ebuf  = xrbuf + (long)NMAX * 64;
  float* wbuf  = ebuf + Ecnt;
  float* emax  = wbuf + Ecnt;
  float* wsum  = emax + NMAX;
  float* sums  = wsum + NMAX;   // 256
  float* gvec  = sums + 256;    // 64
  float* temb  = gvec + 64;     // 64
  float* bsum  = temb + 64;     // 256
  float* csrc  = bsum + 256;    // 64
  float* ctgt  = csrc + 64;     // 64
  float* csec  = ctgt + 64;     // 64

  // ---- 1. per-type encoders: h = relu(x @ W + b), K=16 ----
  for (int t = 0; t < 4; ++t) {
    const float* Wp = (const float*)d_in[PENC + ENC_ORD[t] * 2];
    const float* bp = (const float*)d_in[PENC + ENC_ORD[t] * 2 + 1];
    int blocks = (NSZ[t] + 127) / 128;
    gemm_n64<16, 1><<<blocks, 256, 0, stream>>>(X[t], Wp, bp, hbuf + HOFF[t], NSZ[t]);
  }

  // ---- 2. two HeteroConv(GATv2) layers ----
  for (int l = 0; l < 2; ++l) {
    hipMemsetAsync(nbuf, 0, (size_t)HT * sizeof(float), stream);
    for (int eg = 0; eg < 6; ++eg) {
      int pb = PCONV + l * 36 + eg * 6; // Wl, Wr, att, bias, bl, br
      const float* Wl  = (const float*)d_in[pb + 0];
      const float* Wr  = (const float*)d_in[pb + 1];
      const float* att = (const float*)d_in[pb + 2];
      const float* bl  = (const float*)d_in[pb + 4];
      const float* br  = (const float*)d_in[pb + 5];
      const int*   ei  = (const int*)d_in[EG_EI[eg]];
      int st = EG_SRC[eg], dt = EG_DST[eg];
      int Nsrc = NSZ[st], Ndst = NSZ[dt];

      gemm_n64<64, 0><<<(Nsrc + 127) / 128, 256, 0, stream>>>(hbuf + HOFF[st], Wl, bl, xlbuf, Nsrc);
      gemm_n64<64, 0><<<(Ndst + 127) / 128, 256, 0, stream>>>(hbuf + HOFF[dt], Wr, br, xrbuf, Ndst);

      hipMemsetAsync(emax, 0xFF, (size_t)Ndst * sizeof(float), stream); // -inf-like sentinel
      hipMemsetAsync(wsum, 0,    (size_t)Ndst * sizeof(float), stream);

      long t8 = (long)Ecnt * 8, t16 = (long)Ecnt * 16;
      edge_logits<<<(unsigned)((t8 + 255) / 256), 256, 0, stream>>>(xlbuf, xrbuf, ei, att, ebuf, emax, Ecnt);
      edge_softmax_num<<<(Ecnt + 255) / 256, 256, 0, stream>>>(ebuf, emax, ei, wbuf, wsum, Ecnt);
      edge_scatter<<<(unsigned)((t16 + 255) / 256), 256, 0, stream>>>(xlbuf, ei, wbuf, wsum,
                                                                      nbuf + HOFF[dt], Ecnt);
    }
    // fold per-edge-type GATv2 biases (summed per destination type) into relu pass
    const float* B[6];
    for (int eg = 0; eg < 6; ++eg) B[eg] = (const float*)d_in[PCONV + l * 36 + eg * 6 + 3];
    vec_accum3<<<1, 64, 0, stream>>>(B[1], nullptr, nullptr, bsum + 0 * 64); // network <- host_in_network
    vec_accum3<<<1, 64, 0, stream>>>(B[0], B[4], B[5],       bsum + 1 * 64); // host <- 3 edge types
    vec_accum3<<<1, 64, 0, stream>>>(B[2], nullptr, nullptr, bsum + 2 * 64); // service
    vec_accum3<<<1, 64, 0, stream>>>(B[3], nullptr, nullptr, bsum + 3 * 64); // data
    for (int t = 0; t < 4; ++t) {
      long n = (long)NSZ[t] * 64;
      add_bias_relu<<<(unsigned)((n + 255) / 256), 256, 0, stream>>>(nbuf + HOFF[t], bsum + t * 64,
                                                                     hbuf + HOFF[t], n);
    }
  }

  // ---- 3. global mean pooling + type embedding ----
  hipMemsetAsync(sums, 0, 256 * sizeof(float), stream);
  for (int t = 0; t < 4; ++t)
    colsum<<<1024, 256, 0, stream>>>(hbuf + HOFF[t], NSZ[t], sums + t * 64);
  finalize_g<<<1, 64, 0, stream>>>(sums, (const float*)d_in[PTEMB], (const int*)d_in[PTIDX],
                                   gvec, temb, NSZ[0], NSZ[1], NSZ[2], NSZ[3]);

  float* out = (float*)d_out;

  // ---- 4a. type logits + value (ctx = [g, state_summary]) ----
  ctx_heads<<<1, 64, 0, stream>>>(gvec, (const float*)d_in[PSS],
      (const float*)d_in[PTYPE + 0], (const float*)d_in[PTYPE + 1],
      (const float*)d_in[PTYPE + 2], (const float*)d_in[PTYPE + 3],
      (const float*)d_in[PVAL + 0],  (const float*)d_in[PVAL + 1],
      (const float*)d_in[PVAL + 2],  (const float*)d_in[PVAL + 3], out);

  // ---- 4b. per-node heads: constant-part + fused WMMA GEMM/dot ----
  const float* srcemb = hbuf + HOFF[1]; // h['host'][0]
  const float* tgtemb = hbuf + HOFF[3]; // h['data'][0]

  cbias<<<1, 64, 0, stream>>>((const float*)d_in[PSRC + 0], (const float*)d_in[PSRC + 2],
                              gvec, temb, nullptr, nullptr, csrc);
  head_score<<<(NSZ[1] + 127) / 128, 256, 0, stream>>>(hbuf + HOFF[1], (const float*)d_in[PSRC + 0],
      csrc, (const float*)d_in[PSRC + 1], (const float*)d_in[PSRC + 3], out + 6, NSZ[1]);

  cbias<<<1, 64, 0, stream>>>((const float*)d_in[PTGT + 0], (const float*)d_in[PTGT + 2],
                              gvec, temb, srcemb, nullptr, ctgt);
  head_score<<<(NSZ[3] + 127) / 128, 256, 0, stream>>>(hbuf + HOFF[3], (const float*)d_in[PTGT + 0],
      ctgt, (const float*)d_in[PTGT + 1], (const float*)d_in[PTGT + 3], out + 6 + NSZ[1], NSZ[3]);

  cbias<<<1, 64, 0, stream>>>((const float*)d_in[PSEC + 0], (const float*)d_in[PSEC + 2],
                              gvec, temb, srcemb, tgtemb, csec);
  head_score<<<(NSZ[1] + 127) / 128, 256, 0, stream>>>(hbuf + HOFF[1], (const float*)d_in[PSEC + 0],
      csec, (const float*)d_in[PSEC + 1], (const float*)d_in[PSEC + 3],
      out + 6 + NSZ[1] + NSZ[3], NSZ[1]);
}
